// CameraPedRescue_34729105555636
// MI455X (gfx1250) — matrix operand order repaired and checked by reading, not compiled, not run on hardware
//
#include <hip/hip_runtime.h>
#include <hip/hip_bf16.h>
#include <math.h>

typedef __attribute__((ext_vector_type(16))) _Float16 v16h;
typedef __attribute__((ext_vector_type(8)))  _Float16 v8h;
typedef __attribute__((ext_vector_type(8)))  float    v8f;

#define BEV_HW 512
#define FH 64
#define FW 176
#define NPIX_F (4*FH*FW)          // 45056 feature pixels
#define TOPK 200
#define BN_EPS 1e-5f

// ---------------------------------------------------------------------------
// Weight repack: OIHW f32 -> [Cout][K] f16 with k = (ky*3+kx)*Cin + ci
// ---------------------------------------------------------------------------
__global__ void pack_weights_kernel(const float* __restrict__ w,
                                    _Float16* __restrict__ out,
                                    int Cin, int Cout) {
  int K = Cin * 9;
  int idx = blockIdx.x * blockDim.x + threadIdx.x;
  if (idx >= Cout * K) return;
  int co   = idx / K;
  int k    = idx % K;
  int kpos = k / Cin;      // ky*3+kx
  int ci   = k % Cin;
  out[idx] = (_Float16)w[(co * Cin + ci) * 9 + kpos];
}

// ---------------------------------------------------------------------------
// Conv1: 3->32, 3x3 s2 p1, direct (K=27 too skinny for WMMA). NCHW f32 in,
// NHWC f16 out, fused BN+ReLU. Weights cached in LDS.
// ---------------------------------------------------------------------------
__global__ void conv1_direct_kernel(const float* __restrict__ img,
                                    const float* __restrict__ W1,
                                    const float* __restrict__ g,
                                    const float* __restrict__ bb,
                                    _Float16* __restrict__ out) {
  __shared__ float w[32 * 27];
  for (int i = threadIdx.x; i < 32 * 27; i += blockDim.x) w[i] = W1[i];
  __syncthreads();

  const int OH = 256, OW = 704, IH = 512, IW = 1408;
  int pix = blockIdx.x * blockDim.x + threadIdx.x;
  if (pix >= 4 * OH * OW) return;
  int b   = pix / (OH * OW);
  int rem = pix % (OH * OW);
  int oy = rem / OW, ox = rem % OW;

  float in27[27];
#pragma unroll
  for (int ci = 0; ci < 3; ++ci)
#pragma unroll
    for (int ky = 0; ky < 3; ++ky)
#pragma unroll
      for (int kx = 0; kx < 3; ++kx) {
        int ih = oy * 2 - 1 + ky;
        int iw = ox * 2 - 1 + kx;
        float v = 0.0f;
        if (ih >= 0 && ih < IH && iw >= 0 && iw < IW)
          v = img[(((size_t)b * 3 + ci) * IH + ih) * IW + iw];
        in27[ci * 9 + ky * 3 + kx] = v;
      }

  const float rs = rsqrtf(1.0f + BN_EPS);
  _Float16* op = out + (size_t)pix * 32;
#pragma unroll
  for (int co = 0; co < 32; ++co) {
    float acc = 0.0f;
#pragma unroll
    for (int j = 0; j < 27; ++j) acc = fmaf(in27[j], w[co * 27 + j], acc);
    float val = fmaxf(acc * (g[co] * rs) + bb[co], 0.0f);
    op[co] = (_Float16)val;
  }
}

// ---------------------------------------------------------------------------
// 3x3 conv as implicit GEMM with WMMA f16 (f32 accumulate), specialized at
// compile time on (CIN, COUT, STRIDE); pad = 1.
//   in  : NHWC f16 activations
//   wcol: [COUT][K] f16, k = kpos*CIN + ci (each 32-K chunk is one tap)
//   out : NHWC f16, fused BN (g*rsqrt(1+eps), bias) + ReLU
//
// MI455X data path:
//   * The packed weight matrix (36-162 KB, fits in the 320 KB WGP LDS) is
//     cooperatively staged into dynamic LDS with GLOBAL_LOAD_ASYNC_TO_LDS_B128
//     (ASYNCcnt), fenced with s_wait_asynccnt + barrier. Every B fragment then
//     comes from LDS (ds_load_b128) instead of hammering L2 per wave.
//   * One wave -> 16 pixels x COUT channels: the A fragment is loaded once per
//     32-K chunk and reused across COUT/16 back-to-back WMMAs.
//   * Fully unrolled: 9 taps * CIN/32 chunks * COUT/16 static WMMA ops.
// ---------------------------------------------------------------------------
template <int CIN, int COUT, int STRIDE>
__global__ void conv_wmma_kernel(const _Float16* __restrict__ in,
                                 const _Float16* __restrict__ wcol,
                                 const float* __restrict__ g,
                                 const float* __restrict__ bb,
                                 _Float16* __restrict__ out,
                                 int B, int IH, int IW, int OH, int OW) {
  constexpr int K      = CIN * 9;
  constexpr int NT     = COUT / 16;      // accumulator tiles along channels
  constexpr int WBYTES = COUT * K * 2;   // packed weight bytes (mult of 16)

  extern __shared__ _Float16 smemW[];    // dynamic LDS: whole weight matrix

  // ---- cooperative async stage of weights into LDS (all threads) ----
  {
    const unsigned ldsBase = __builtin_amdgcn_groupstaticsize(); // dyn-LDS base
    const char* gsrc = (const char*)wcol;
    for (int off = (int)threadIdx.x * 16; off < WBYTES; off += 256 * 16) {
      unsigned long long ga = (unsigned long long)(gsrc + off);
      unsigned la = ldsBase + (unsigned)off;
      asm volatile("global_load_async_to_lds_b128 %0, %1, off"
                   :: "v"(la), "v"(ga) : "memory");
    }
    asm volatile("s_wait_asynccnt 0x0" ::: "memory");
    __syncthreads();
  }

  const int Mtiles = (B * OH * OW) >> 4;
  int wave = (blockIdx.x * blockDim.x + threadIdx.x) >> 5;
  int lane = threadIdx.x & 31;
  if (wave >= Mtiles) return;            // wave-uniform: EXEC all-ones at WMMA
  const int mtile = wave;                // no division: one wave per pixel tile

  const int upper = lane >> 4;           // 0: lanes 0-15, 1: lanes 16-31
  // --- A fragment addressing: row M = lane&15 ---
  int mrow = lane & 15;
  int pix  = mtile * 16 + mrow;
  int b    = pix / (OH * OW);
  int rem  = pix % (OH * OW);
  int oy = rem / OW, ox = rem % OW;
  const int kbA = upper ? 8 : 0;         // K offset of this lane's A halves

  // --- B fragment addressing: col N = lane&15, K halves at kbA and 16+kbA (in LDS) ---
  int ncol = lane & 15;
  const _Float16* wbase = smemW + (size_t)ncol * K + (upper ? 8 : 0);

  v8f acc[NT];
#pragma unroll
  for (int nt = 0; nt < NT; ++nt) acc[nt] = (v8f){};

#pragma unroll
  for (int kpos = 0; kpos < 9; ++kpos) {
    constexpr int PAD = 1;
    const int ky = kpos / 3, kx = kpos % 3;
    const int ih = oy * STRIDE - PAD + ky;
    const int iw = ox * STRIDE - PAD + kx;
    const bool inb = (ih >= 0) && (ih < IH) && (iw >= 0) && (iw < IW);
    const _Float16* ap =
        in + (((size_t)b * IH + ih) * IW + iw) * CIN + kbA;

#pragma unroll
    for (int c = 0; c < CIN / 32; ++c) {
      v8h a0 = (v8h){}, a1 = (v8h){};
      if (inb) {
        a0 = *(const v8h*)(ap + c * 32);        // K = kk + kbA .. +7
        a1 = *(const v8h*)(ap + c * 32 + 16);   // K = kk + 16 + kbA .. +7
      }
      v16h amat = __builtin_shufflevector(a0, a1,
          0, 1, 2, 3, 4, 5, 6, 7, 8, 9, 10, 11, 12, 13, 14, 15);

      const int kk = kpos * CIN + c * 32;
#pragma unroll
      for (int nt = 0; nt < NT; ++nt) {
        const _Float16* wp = wbase + (size_t)nt * 16 * K + kk;
        v8h b0 = *(const v8h*)(wp);             // ds_load_b128
        v8h b1 = *(const v8h*)(wp + 16);        // ds_load_b128
        v16h bmat = __builtin_shufflevector(b0, b1,
            0, 1, 2, 3, 4, 5, 6, 7, 8, 9, 10, 11, 12, 13, 14, 15);
        acc[nt] = __builtin_amdgcn_wmma_f32_16x16x32_f16(
            /*neg_a=*/false, amat, /*neg_b=*/false, bmat,
            /*c_mod=*/(short)0, acc[nt], /*reuse_a=*/false, /*reuse_b=*/false);
      }
    }
  }

  // --- epilogue: BN + ReLU, store f16 NHWC ---
  const float rs = rsqrtf(1.0f + BN_EPS);
  const int mbase = upper ? 8 : 0;
#pragma unroll
  for (int nt = 0; nt < NT; ++nt) {
    int nglob   = nt * 16 + ncol;
    float scale = g[nglob] * rs;
    float bias  = bb[nglob];
#pragma unroll
    for (int v = 0; v < 8; ++v) {
      int opix  = mtile * 16 + mbase + v;
      float val = fmaxf(acc[nt][v] * scale + bias, 0.0f);
      out[(size_t)opix * COUT + nglob] = (_Float16)val;
    }
  }
}

// ---------------------------------------------------------------------------
// 1x1 head conv (96 -> 1) + bias.
// mode 0 (heatmap): out_lin = raw logit, out_prob = sigmoid(logit)
// mode 1 (depth)  : out_lin = DMIN + sigmoid(logit)*(DMAX-DMIN)
// ---------------------------------------------------------------------------
__global__ void head1x1_kernel(const _Float16* __restrict__ hid,
                               const float* __restrict__ w,
                               const float* __restrict__ bias,
                               float* __restrict__ out_lin,
                               float* __restrict__ out_prob,
                               int mode) {
  int pix = blockIdx.x * blockDim.x + threadIdx.x;
  if (pix >= NPIX_F) return;
  const _Float16* h = hid + (size_t)pix * 96;
  float s = bias[0];
#pragma unroll
  for (int j = 0; j < 96; ++j) s = fmaf((float)h[j], w[j], s);
  if (mode == 0) {
    out_lin[pix]  = s;
    out_prob[pix] = 1.0f / (1.0f + expf(-s));
  } else {
    out_lin[pix]  = 1.0f + 79.0f / (1.0f + expf(-s));   // DMIN + sig*(DMAX-DMIN)
  }
}

// ---------------------------------------------------------------------------
// Per-batch top-200 by iterative argmax in LDS (ties -> lowest index, like
// lax.top_k). One workgroup per batch; 11264 floats resident in LDS.
// ---------------------------------------------------------------------------
__global__ void topk_kernel(const float* __restrict__ probs,
                            int* __restrict__ tidx,
                            float* __restrict__ tscore) {
  const int N = FH * FW;              // 11264
  __shared__ float sv[FH * FW];
  __shared__ float rmax[256];
  __shared__ int   ridx[256];
  int b   = blockIdx.x;
  int tid = threadIdx.x;
  const float* p = probs + (size_t)b * N;
  for (int i = tid; i < N; i += 256) sv[i] = p[i];
  __syncthreads();

  for (int k = 0; k < TOPK; ++k) {
    float best = -1.0f;
    int   bi   = 0x7fffffff;
    for (int i = tid; i < N; i += 256) {
      float v = sv[i];
      if (v > best || (v == best && i < bi)) { best = v; bi = i; }
    }
    rmax[tid] = best; ridx[tid] = bi;
    __syncthreads();
    for (int s = 128; s > 0; s >>= 1) {
      if (tid < s) {
        if (rmax[tid + s] > rmax[tid] ||
            (rmax[tid + s] == rmax[tid] && ridx[tid + s] < ridx[tid])) {
          rmax[tid] = rmax[tid + s];
          ridx[tid] = ridx[tid + s];
        }
      }
      __syncthreads();
    }
    if (tid == 0) {
      tidx[b * TOPK + k]   = ridx[0];
      tscore[b * TOPK + k] = rmax[0];
      sv[ridx[0]] = -2.0f;            // remove from further consideration
    }
    __syncthreads();
  }
}

// ---------------------------------------------------------------------------
// Zero the BEV accumulator (bit pattern 0 == 0.0f, valid for atomicMax trick)
// ---------------------------------------------------------------------------
__global__ void zero_kernel(float* __restrict__ buf, int n) {
  int i = blockIdx.x * blockDim.x + threadIdx.x;
  if (i < n) buf[i] = 0.0f;
}

// ---------------------------------------------------------------------------
// Project peaks camera->lidar->BEV grid, splat 5x5 Gaussian with max-combine.
// Scores/gauss are >= 0 so float max == uint max on raw bits.
// ---------------------------------------------------------------------------
__global__ void splat_kernel(const int* __restrict__ tidx,
                             const float* __restrict__ tscore,
                             const float* __restrict__ depth,   // [4][FH][FW]
                             const float* __restrict__ P,       // [4][3][4]
                             const float* __restrict__ T,       // [4][4][4]
                             float* __restrict__ bev) {         // [4][512][512]
  int t = blockIdx.x * blockDim.x + threadIdx.x;
  if (t >= 4 * TOPK) return;
  int b     = t / TOPK;
  float sc  = tscore[t];
  int idx   = tidx[t];
  int ys = idx / FW, xs = idx % FW;
  float d = depth[(size_t)b * (FH * FW) + idx];
  float u = ((float)xs + 0.5f) * (1408.0f / (float)FW);
  float v = ((float)ys + 0.5f) * (512.0f / (float)FH);

  const float* Pb = P + b * 12;
  float fx = fmaxf(Pb[0], 1e-4f), fy = fmaxf(Pb[5], 1e-4f);
  float cx = Pb[2], cy = Pb[6];
  float tx = Pb[3], ty = Pb[7];
  float xc = (u * d - cx * d - tx) / fx;
  float yc = (v * d - cy * d - ty) / fy;

  const float* Tb = T + b * 16;
  float lx = Tb[0] * xc + Tb[1] * yc + Tb[2] * d + Tb[3];
  float ly = Tb[4] * xc + Tb[5] * yc + Tb[6] * d + Tb[7];

  int gx = (int)floorf((lx + 51.2f) / 0.2f);
  int gy = (int)floorf((ly + 51.2f) / 0.2f);
  bool valid = (sc >= 0.15f) && gx >= 0 && gx < BEV_HW && gy >= 0 && gy < BEV_HW;
  if (!valid) return;

  // sigma = 5/6 -> exponent = -r^2 * (1/(2*sigma^2)) = -r^2 * 0.72
#pragma unroll
  for (int dy = -2; dy <= 2; ++dy) {
#pragma unroll
    for (int dx = -2; dx <= 2; ++dx) {
      int iy = gy + dy, ix = gx + dx;
      if (iy < 0 || iy >= BEV_HW || ix < 0 || ix >= BEV_HW) continue;
      float val = sc * expf(-(float)(dy * dy + dx * dx) * 0.72f);
      unsigned* addr =
          (unsigned*)&bev[((size_t)b * BEV_HW + iy) * BEV_HW + ix];
      atomicMax(addr, __float_as_uint(val));
    }
  }
}

// ---------------------------------------------------------------------------
// bev -> clipped prob + logits
// ---------------------------------------------------------------------------
__global__ void finalize_kernel(float* __restrict__ bev_logits,
                                float* __restrict__ bev_prob, int n) {
  int i = blockIdx.x * blockDim.x + threadIdx.x;
  if (i >= n) return;
  float v = bev_prob[i];
  float p = fminf(fmaxf(v, 1e-4f), 1.0f - 1e-4f);
  bev_prob[i]   = p;
  bev_logits[i] = logf(p) - log1pf(-p);
}

// ===========================================================================
extern "C" void kernel_launch(void* const* d_in, const int* in_sizes, int n_in,
                              void* d_out, int out_size, void* d_ws, size_t ws_size,
                              hipStream_t stream) {
  (void)in_sizes; (void)n_in; (void)out_size; (void)ws_size;

  const float* image = (const float*)d_in[0];   // [4,3,512,1408]
  const float* camP  = (const float*)d_in[1];   // [4,3,4]
  const float* Tlc   = (const float*)d_in[2];   // [4,4,4]
  const float* W1  = (const float*)d_in[3];
  const float* g1  = (const float*)d_in[4];
  const float* b1  = (const float*)d_in[5];
  const float* W2  = (const float*)d_in[6];
  const float* g2  = (const float*)d_in[7];
  const float* b2  = (const float*)d_in[8];
  const float* W3  = (const float*)d_in[9];
  const float* g3  = (const float*)d_in[10];
  const float* b3  = (const float*)d_in[11];
  const float* Wh1 = (const float*)d_in[12];
  const float* gh1 = (const float*)d_in[13];
  const float* bh1 = (const float*)d_in[14];
  const float* Wh2 = (const float*)d_in[15];
  const float* bh2 = (const float*)d_in[16];
  const float* Wd1 = (const float*)d_in[17];
  const float* gd1 = (const float*)d_in[18];
  const float* bd1 = (const float*)d_in[19];
  const float* Wd2 = (const float*)d_in[20];
  const float* bd2 = (const float*)d_in[21];

  // ---- d_out layout (f32): hm | depth | bev_logits | bev_prob ----
  float* out_hm     = (float*)d_out;                    // 45056
  float* out_depth  = out_hm + NPIX_F;                  // 45056
  float* out_logits = out_depth + NPIX_F;               // 4*512*512
  float* out_prob   = out_logits + 4 * BEV_HW * BEV_HW; // 4*512*512 (also accum)

  // ---- workspace carve-up (256B aligned) ----
  char* ws = (char*)d_ws;
  size_t off = 0;
  auto alloc = [&](size_t bytes) -> char* {
    char* p = ws + off;
    off += (bytes + 255) & ~(size_t)255;
    return p;
  };
  _Float16* act1 = (_Float16*)alloc((size_t)4 * 256 * 704 * 32 * 2); // 46.1 MB
  _Float16* act2 = (_Float16*)alloc((size_t)4 * 128 * 352 * 64 * 2); // 23.1 MB
  _Float16* act3 = (_Float16*)alloc((size_t)NPIX_F * 96 * 2);        //  8.7 MB
  _Float16* hidH = (_Float16*)alloc((size_t)NPIX_F * 96 * 2);
  _Float16* hidD = (_Float16*)alloc((size_t)NPIX_F * 96 * 2);
  _Float16* W2c  = (_Float16*)alloc((size_t)64 * 288 * 2);
  _Float16* W3c  = (_Float16*)alloc((size_t)96 * 576 * 2);
  _Float16* Wh1c = (_Float16*)alloc((size_t)96 * 864 * 2);
  _Float16* Wd1c = (_Float16*)alloc((size_t)96 * 864 * 2);
  float*    probs  = (float*)alloc((size_t)4 * FH * FW * 4);
  int*      tidx   = (int*)alloc((size_t)4 * TOPK * 4);
  float*    tscore = (float*)alloc((size_t)4 * TOPK * 4);

  const int TB = 256;
  auto blocks = [](long n, int tb) { return (int)((n + tb - 1) / tb); };

  // 1) repack weights to [Cout][K] f16
  pack_weights_kernel<<<blocks(64 * 288, TB), TB, 0, stream>>>(W2, W2c, 32, 64);
  pack_weights_kernel<<<blocks(96 * 576, TB), TB, 0, stream>>>(W3, W3c, 64, 96);
  pack_weights_kernel<<<blocks(96 * 864, TB), TB, 0, stream>>>(Wh1, Wh1c, 96, 96);
  pack_weights_kernel<<<blocks(96 * 864, TB), TB, 0, stream>>>(Wd1, Wd1c, 96, 96);

  // 2) conv1 (direct) -> act1 NHWC f16
  conv1_direct_kernel<<<blocks((long)4 * 256 * 704, TB), TB, 0, stream>>>(
      image, W1, g1, b1, act1);

  // 3) conv2 WMMA: [4,256,704,32] -> [4,128,352,64], s2 p1; weights in LDS
  {
    long waves = (long)4 * 128 * 352 / 16;               // 11264 pixel tiles
    conv_wmma_kernel<32, 64, 2>
        <<<blocks(waves * 32, TB), TB, 64 * 288 * 2, stream>>>(
        act1, W2c, g2, b2, act2, 4, 256, 704, 128, 352);
  }
  // 4) conv3 WMMA: -> [4,64,176,96], s2 p1; weights in LDS (108 KB)
  {
    long waves = (long)NPIX_F / 16;                      // 2816 pixel tiles
    conv_wmma_kernel<64, 96, 2>
        <<<blocks(waves * 32, TB), TB, 96 * 576 * 2, stream>>>(
        act2, W3c, g3, b3, act3, 4, 128, 352, FH, FW);
  }
  // 5) head hidden convs WMMA: 96->96, s1 p1; weights in LDS (162 KB)
  {
    long waves = (long)NPIX_F / 16;
    conv_wmma_kernel<96, 96, 1>
        <<<blocks(waves * 32, TB), TB, 96 * 864 * 2, stream>>>(
        act3, Wh1c, gh1, bh1, hidH, 4, FH, FW, FH, FW);
    conv_wmma_kernel<96, 96, 1>
        <<<blocks(waves * 32, TB), TB, 96 * 864 * 2, stream>>>(
        act3, Wd1c, gd1, bd1, hidD, 4, FH, FW, FH, FW);
  }

  // 6) 1x1 heads: hm logits + sigmoid probs; depth map
  head1x1_kernel<<<blocks(NPIX_F, TB), TB, 0, stream>>>(
      hidH, Wh2, bh2, out_hm, probs, 0);
  head1x1_kernel<<<blocks(NPIX_F, TB), TB, 0, stream>>>(
      hidD, Wd2, bd2, out_depth, probs, 1);

  // 7) top-200 per batch
  topk_kernel<<<4, 256, 0, stream>>>(probs, tidx, tscore);

  // 8) BEV accumulate (max-splat) then logits/prob
  int nbev = 4 * BEV_HW * BEV_HW;
  zero_kernel<<<blocks(nbev, TB), TB, 0, stream>>>(out_prob, nbev);
  splat_kernel<<<blocks(4 * TOPK, TB), TB, 0, stream>>>(
      tidx, tscore, out_depth, camP, Tlc, out_prob);
  finalize_kernel<<<blocks(nbev, TB), TB, 0, stream>>>(out_logits, out_prob, nbev);
}
